// TreeLSTMCell_13134009991193
// MI455X (gfx1250) — compile-verified, matrix operand order checked by reading, not run on hardware
//
#include <hip/hip_runtime.h>
#include <hip/hip_bf16.h>
#include <stdint.h>

// ---------------------------------------------------------------------------
// TreeLSTM reduce batch fused as one GEMM + epilogue.
//   A  : [N, 1024] = [x_left | x_right | h_child0 | h_child1] (bf16, in LDS)
//   Wc : [1024 out, 1024 in] bf16 combined weights (built once in d_ws):
//        rows   0..255  i gate | 256..511 o | 512..767 u |
//        rows 768..1023 f gate with U_f folded: U_f_W[j,:]+U_f_W[j+256,:]
//   epilogue : c = sig(i)*tanh(u) + sig(f)*(c0+c1);  h = sig(o)*tanh(c)
//
// Per workgroup (128 threads = 4 waves): 128-node M tile.
//   - A tile fp32->bf16 in LDS, XOR bank-swizzled (256 KB)
//   - B slices (64 rows x 256 K bf16 = 32 KB) DOUBLE-BUFFERED via
//     global_load_async_to_lds_b128: slice s+1 DMA overlaps slice s compute
//   - each wave owns TWO 16-row strips -> 1.5 ds_load_b128 per v_wmma
//   - LDS = 256 KB A + 2x32 KB B = 320 KB (per-workgroup max; WGP$ separate)
// ---------------------------------------------------------------------------

typedef __attribute__((ext_vector_type(16))) __bf16 v16bf;
typedef __attribute__((ext_vector_type(8)))  float  v8f;
typedef __attribute__((ext_vector_type(4)))  float  v4f;   // clang vector: OK for NT builtins

#define NNODES     131072
#define HDIM       256
#define KDIM       1024
#define MTILE      128
#define ABYTES_ROW 2048                      // 1024 bf16 per A row
#define BBYTES_ROW 512                       // 256 bf16 per staged B row
#define ABUF_BYTES (MTILE * ABYTES_ROW)      // 262144
#define BBUF_BYTES (64 * BBYTES_ROW)         // 32768 per buffer, x2 -> 320 KB total

__device__ __forceinline__ float sigf(float x) {
  return 1.0f / (1.0f + __expf(-x));
}
__device__ __forceinline__ float tanh_fast(float x) {
  return 2.0f / (1.0f + __expf(-2.0f * x)) - 1.0f;
}

// XOR-swizzled LDS addresses: 32B chunks, chunk ^= (row & 7).
// Power-of-two row strides would alias all fragment rows onto one bank group;
// the swizzle spreads a 16-row fragment over 8 groups, fragments stay 32B aligned.
__device__ __forceinline__ uint32_t a_addr(int row, int kelem) {
  int c = kelem >> 4;
  return (uint32_t)(row * ABYTES_ROW + (((c ^ (row & 7)) << 5) | ((kelem & 15) << 1)));
}
__device__ __forceinline__ uint32_t b_addr(int buf, int row, int kelem) {  // kelem in [0,256)
  int c = kelem >> 4;
  return (uint32_t)(ABUF_BYTES + buf * BBUF_BYTES + row * BBYTES_ROW +
                    (((c ^ (row & 7)) << 5) | ((kelem & 15) << 1)));
}

__device__ __forceinline__ void stage4(char* smem, int row, int kelem, const float* src) {
  v4f v = __builtin_nontemporal_load((const v4f*)src);
  __bf16* p = (__bf16*)(smem + a_addr(row, kelem));
  p[0] = (__bf16)v.x; p[1] = (__bf16)v.y; p[2] = (__bf16)v.z; p[3] = (__bf16)v.w;
}

// Async-DMA one B slice (h-block hb, K-quarter q) into LDS buffer `buf`.
// 64 rows x 512B = 2048 x 16B units, 16 issues per thread (128 threads).
__device__ __forceinline__ void issue_b_slice(const __bf16* Wc, uint32_t lds0,
                                              int hb, int q, int buf, int tid) {
  #pragma unroll 4
  for (int it = 0; it < 16; ++it) {
    int u      = it * 128 + tid;          // 16B unit id
    int rIdx   = u >> 5;                  // 0..63 (gate*16 + r)
    int wu     = u & 31;                  // 16B unit within row
    int srcrow = ((rIdx >> 4) << 8) + hb * 16 + (rIdx & 15);
    unsigned long long g64 = (unsigned long long)
        ((const char*)Wc + (size_t)srcrow * (KDIM * 2) + q * BBYTES_ROW + wu * 16);
    uint32_t dst = lds0 + (uint32_t)(ABUF_BYTES + buf * BBUF_BYTES + rIdx * BBYTES_ROW +
                   ((((wu >> 1) ^ (rIdx & 7)) << 5) | ((wu & 1) << 4)));
    asm volatile("global_load_async_to_lds_b128 %0, %1, off"
                 :: "v"(dst), "v"(g64) : "memory");
  }
}

// ---------------- weight prep: fold + convert to bf16 ----------------------
__global__ void treelstm_prep(const float* __restrict__ Wl,  const float* __restrict__ Wr,
                              const float* __restrict__ Fl,  const float* __restrict__ Fr,
                              const float* __restrict__ Uiou,const float* __restrict__ UfW,
                              const float* __restrict__ Ufb, const float* __restrict__ biou,
                              __bf16* __restrict__ Wc, float* __restrict__ bias) {
  int idx = blockIdx.x * blockDim.x + threadIdx.x;   // 0 .. 1024*1024-1
  int r = idx >> 10;
  int k = idx & 1023;
  float w;
  if (r < 768) {                       // i, o, u gates
    if (k < 256)       w = Wl[r * 256 + k];
    else if (k < 512)  w = Wr[r * 256 + (k - 256)];
    else               w = Uiou[r * 512 + (k - 512)];
  } else {                             // folded forget gate
    int j = r - 768;
    if (k < 256)       w = Fl[j * 256 + k];
    else if (k < 512)  w = Fr[j * 256 + (k - 256)];
    else               w = UfW[j * 512 + (k - 512)] + UfW[(j + 256) * 512 + (k - 512)];
  }
  Wc[idx] = (__bf16)w;
  if (idx < 1024) {
    bias[idx] = (idx < 768) ? biou[idx] : (Ufb[idx - 768] + Ufb[idx - 768 + 256]);
  }
}

// ---------------- fused GEMM + LSTM epilogue -------------------------------
__global__ void __launch_bounds__(128, 1)
treelstm_gemm(const float* __restrict__ x_left,  const float* __restrict__ x_right,
              const float* __restrict__ child_h, const float* __restrict__ child_c,
              const __bf16* __restrict__ Wc,     const float* __restrict__ bias,
              float* __restrict__ out) {
  extern __shared__ char smem[];

  const int tid  = threadIdx.x;                 // 128 threads = 4 waves (wave32)
  const int lane = tid & 31;
  const int wave = tid >> 5;
  const long nodeBase = (long)blockIdx.x * MTILE;

  // ---- stage A tile: fp32 -> bf16 into swizzled LDS (1 row / thread) ------
  {
    const int  m    = tid;
    const long node = nodeBase + m;
    #pragma unroll 4
    for (int q = 0; q < HDIM; q += 4)           // K 0..255 : x_left
      stage4(smem, m, q, x_left + node * HDIM + q);
    #pragma unroll 4
    for (int q = 0; q < HDIM; q += 4)           // K 256..511 : x_right
      stage4(smem, m, HDIM + q, x_right + node * HDIM + q);
    #pragma unroll 4
    for (int q = 0; q < 2 * HDIM; q += 4)       // K 512..1023 : child_h (both kids)
      stage4(smem, m, 512 + q, child_h + node * 2 * HDIM + q);
  }

  const int rsel  = lane & 15;
  const int kselE = (lane >> 4) * 16;           // K-half select inside fragment
  const int s0    = wave * 32;                  // this wave's two strips: s0, s0+16
  const uint32_t lds0 = (uint32_t)(uintptr_t)smem;

  // ---- software-pipelined B stream: 64 slices = (hb 0..15) x (quarter 0..3)
  issue_b_slice(Wc, lds0, 0, 0, 0, tid);        // prologue: slice 0 -> buf 0

  v8f acc[2][4];
  for (int s = 0; s < 64; ++s) {
    const int hb  = s >> 2;
    const int q   = s & 3;
    const int buf = s & 1;

    if (q == 0) {
      #pragma unroll
      for (int t = 0; t < 2; ++t)
        #pragma unroll
        for (int g = 0; g < 4; ++g)
          acc[t][g] = (v8f){0.f,0.f,0.f,0.f,0.f,0.f,0.f,0.f};
    }

    // slice s (issued last iteration) has landed in buf; barrier also proves
    // every wave is done reading buf^1 (slice s-1), which slice s+1 overwrites.
    asm volatile("s_wait_asynccnt 0" ::: "memory");
    __syncthreads();
    if (s < 63)
      issue_b_slice(Wc, lds0, (s + 1) >> 2, (s + 1) & 3, (s + 1) & 1, tid);

    // ---- K accumulation: A frags shared across 4 gates, B across 2 strips -
    #pragma unroll
    for (int kk = 0; kk < 8; ++kk) {
      const int ke  = q * 256 + kk * 32 + kselE;   // element offset in A row
      const int keB = kk * 32 + kselE;             // element offset in B slice
      v16bf a0 = *(const v16bf*)(smem + a_addr(s0 +      rsel, ke));
      v16bf a1 = *(const v16bf*)(smem + a_addr(s0 + 16 + rsel, ke));
      #pragma unroll
      for (int g = 0; g < 4; ++g) {
        v16bf b = *(const v16bf*)(smem + b_addr(buf, g * 16 + rsel, keB));
        acc[0][g] = __builtin_amdgcn_wmma_f32_16x16x32_bf16(false, a0, false, b,
                                                            (short)0, acc[0][g], false, false);
        acc[1][g] = __builtin_amdgcn_wmma_f32_16x16x32_bf16(false, a1, false, b,
                                                            (short)0, acc[1][g], false, false);
      }
    }

    if (q == 3) {
      // ---- epilogue: node = base + strip + (lane>>4)*8 + v, ch = hb*16+rsel
      const int ch = hb * 16 + rsel;
      const float bI = bias[ch], bO = bias[256 + ch], bU = bias[512 + ch], bF = bias[768 + ch];
      #pragma unroll
      for (int t = 0; t < 2; ++t) {
        const long rowBase = nodeBase + s0 + t * 16 + (lane >> 4) * 8;
        #pragma unroll
        for (int v = 0; v < 8; ++v) {
          long node = rowBase + v;
          float c0 = __builtin_nontemporal_load(child_c + node * 512 + ch);
          float c1 = __builtin_nontemporal_load(child_c + node * 512 + 256 + ch);
          float iv = acc[t][0][v] + bI;
          float ov = acc[t][1][v] + bO;
          float uv = acc[t][2][v] + bU;
          float fv = acc[t][3][v] + bF;
          float c  = sigf(iv) * tanh_fast(uv) + sigf(fv) * (c0 + c1);
          float h  = sigf(ov) * tanh_fast(c);
          __builtin_nontemporal_store(h, out + node * HDIM + ch);
          __builtin_nontemporal_store(c, out + (long)NNODES * HDIM + node * HDIM + ch);
        }
      }
    }
  }
}

// ---------------------------------------------------------------------------
extern "C" void kernel_launch(void* const* d_in, const int* in_sizes, int n_in,
                              void* d_out, int out_size, void* d_ws, size_t ws_size,
                              hipStream_t stream) {
  const float* x_left   = (const float*)d_in[0];
  const float* x_right  = (const float*)d_in[1];
  const float* child_h  = (const float*)d_in[2];
  const float* child_c  = (const float*)d_in[3];
  const float* W_iou_l  = (const float*)d_in[4];
  const float* W_iou_r  = (const float*)d_in[5];
  const float* W_f_l    = (const float*)d_in[6];
  const float* W_f_r    = (const float*)d_in[7];
  const float* U_iou    = (const float*)d_in[8];
  const float* U_f_W    = (const float*)d_in[9];
  const float* U_f_b    = (const float*)d_in[10];
  const float* b_iou    = (const float*)d_in[11];

  // workspace: 2MB bf16 combined weights + 4KB folded bias
  __bf16* Wc  = (__bf16*)d_ws;
  float* bias = (float*)((char*)d_ws + (size_t)KDIM * KDIM * sizeof(__bf16));

  treelstm_prep<<<(KDIM * KDIM) / 256, 256, 0, stream>>>(
      W_iou_l, W_iou_r, W_f_l, W_f_r, U_iou, U_f_W, U_f_b, b_iou, Wc, bias);

  size_t shmem = (size_t)(ABUF_BYTES + 2 * BBUF_BYTES);   // 320 KB (WGP max)
  treelstm_gemm<<<NNODES / MTILE, 128, shmem, stream>>>(
      x_left, x_right, child_h, child_c, Wc, bias, (float*)d_out);
}
